// MultiHeadSeqAttention_90194313216043
// MI455X (gfx1250) — matrix operand order repaired and checked by reading, not compiled, not run on hardware
//
#include <hip/hip_runtime.h>
#include <hip/hip_bf16.h>

// Sizes from reference
#define BB 8
#define MM 512
#define HH 512
#define KK 8      // heads
#define LL 1024   // span
#define DD 64     // head dim
#define NN 1536   // M + L
#define NP 1552   // padded key length (16 extra so masked tiles stay in-bounds)

typedef __attribute__((ext_vector_type(16))) __bf16 v16bf;
typedef __attribute__((ext_vector_type(8)))  float  v8f;

union Frag {
  v16bf v;
  unsigned short us[16];
  uint4 q[2];
};

// Native f32 -> bf16 (RNE) so the backend can use hardware converters
__device__ __forceinline__ unsigned short f2bf(float f) {
  __bf16 b = (__bf16)f;
  return __builtin_bit_cast(unsigned short, b);
}
__device__ __forceinline__ float bf2f(unsigned short s) {
  return __uint_as_float(((unsigned int)s) << 16);
}

// ---- Fragment loaders (ISA 7.12.2 layouts, wave32) ----
// A 16x32 bf16: lane: m = lane&15, h = lane>>4; K chunks [8h,8h+8) and [16+8h,16+8h+8)
__device__ __forceinline__ Frag load_a_bf16(const unsigned short* base, int ld, int lane) {
  const int m = lane & 15, h = lane >> 4;
  Frag f;
  const unsigned short* p = base + m * ld + 8 * h;
  f.q[0] = *(const uint4*)(p);
  f.q[1] = *(const uint4*)(p + 16);
  return f;
}
// B 32x16 bf16: lane: col = lane&15, h = lane>>4; K chunks [8h,8h+8) and [16+8h,16+8h+8)
// 'base' points at row-major storage where row index == output column, stride ld.
__device__ __forceinline__ Frag load_b_bf16(const unsigned short* base, int ld, int lane) {
  const int c = lane & 15, h = lane >> 4;
  Frag f;
  const unsigned short* p = base + c * ld + 8 * h;
  f.q[0] = *(const uint4*)(p);
  f.q[1] = *(const uint4*)(p + 16);
  return f;
}
// A-frag from fp32 source, converted to bf16 on the fly
__device__ __forceinline__ Frag load_a_f32(const float* base, int ld, int lane) {
  const int m = lane & 15, h = lane >> 4;
  Frag f;
  const float* p = base + m * ld + 8 * h;
  float4 x0 = *(const float4*)(p);
  float4 x1 = *(const float4*)(p + 4);
  float4 x2 = *(const float4*)(p + 16);
  float4 x3 = *(const float4*)(p + 20);
  f.us[0]=f2bf(x0.x); f.us[1]=f2bf(x0.y); f.us[2]=f2bf(x0.z); f.us[3]=f2bf(x0.w);
  f.us[4]=f2bf(x1.x); f.us[5]=f2bf(x1.y); f.us[6]=f2bf(x1.z); f.us[7]=f2bf(x1.w);
  f.us[8]=f2bf(x2.x); f.us[9]=f2bf(x2.y); f.us[10]=f2bf(x2.z); f.us[11]=f2bf(x2.w);
  f.us[12]=f2bf(x3.x); f.us[13]=f2bf(x3.y); f.us[14]=f2bf(x3.z); f.us[15]=f2bf(x3.w);
  return f;
}

__device__ __forceinline__ v8f wmma_bf16(Frag a, Frag b, v8f c) {
  return __builtin_amdgcn_wmma_f32_16x16x32_bf16(
      /*neg_a=*/false, a.v, /*neg_b=*/false, b.v,
      /*c_mod=*/(short)0, c, /*reuse_a=*/false, /*reuse_b=*/false);
}

// ---- Workspace layout (ushort elements). Total ~34.4 MB ----
#define WS_WQ  0
#define WS_WK  (WS_WQ + HH*HH)
#define WS_WV  (WS_WK + HH*HH)
#define WS_WO  (WS_WV + HH*HH)
#define WS_PE  (WS_WO + HH*HH)            // peT[l][d], 1024x64
#define WS_QB  (WS_PE + LL*DD)            // q  [64 heads][512][64]
#define WS_KB  (WS_QB + BB*KK*MM*DD)      // k  [64 heads][1552][64]
#define WS_VT  (WS_KB + BB*KK*NP*DD)      // vT [64 heads][64][1552]
#define WS_ATT (WS_VT + BB*KK*NP*DD)      // att heads-concat [8][512][512]

// LDS per wave-tile in attn: qpe(16x1024 bf16) + score/prob(16x1056 bf16)
#define LDS_QPE_B (16 * LL * 2)           // 32768
#define LDS_SCP_B (16 * 1056 * 2)         // 33792
#define LDS_PER_WAVE (LDS_QPE_B + LDS_SCP_B)

// ============ prep: weights fp32->bf16, peT transpose, zero key/value pads ============
__global__ void prep_kernel(const float* __restrict__ Wq, const float* __restrict__ Wk,
                            const float* __restrict__ Wv, const float* __restrict__ Wo,
                            const float* __restrict__ pe, unsigned short* __restrict__ ws) {
  int t = blockIdx.x * blockDim.x + threadIdx.x;
  const int WSZ = HH * HH;
  if (t < WSZ) { ws[WS_WQ + t] = f2bf(Wq[t]); return; } t -= WSZ;
  if (t < WSZ) { ws[WS_WK + t] = f2bf(Wk[t]); return; } t -= WSZ;
  if (t < WSZ) { ws[WS_WV + t] = f2bf(Wv[t]); return; } t -= WSZ;
  if (t < WSZ) { ws[WS_WO + t] = f2bf(Wo[t]); return; } t -= WSZ;
  if (t < LL * DD) {
    int l = t / DD, d = t % DD;
    ws[WS_PE + l * DD + d] = f2bf(pe[d * LL + l]);   // key_pe [D,L] -> peT [L,D]
    return;
  }
  t -= LL * DD;
  const int PADE = BB * KK * (NP - NN) * DD;         // 65536
  if (t < PADE) {
    int head = t / ((NP - NN) * DD);
    int r = t % ((NP - NN) * DD);
    int n = NN + r / DD, d = r % DD;
    ws[WS_KB + (head * NP + n) * DD + d] = 0;
    return;
  }
  t -= PADE;
  if (t < PADE) {
    int head = t / ((NP - NN) * DD);
    int r = t % ((NP - NN) * DD);
    int n = NN + r / DD, d = r % DD;
    ws[WS_VT + (head * DD + d) * NP + n] = 0;
  }
}

// ============ proj: out = X @ W^T, scattered to per-head bf16 layouts ============
// MODE 0: q -> [head][m][d]   MODE 1: k -> [head][n][d]   MODE 2: v -> [head][d][n]
// SEQ is compile-time so token decomposition is mul-hi, not a division expansion.
template <int MODE, int SEQ>
__global__ void __launch_bounds__(128) proj_kernel(const float* __restrict__ X,
                                                   const unsigned short* __restrict__ Wb,
                                                   unsigned short* __restrict__ dst) {
  const int lane = threadIdx.x & 31;
  const int w = (blockIdx.x * blockDim.x + threadIdx.x) >> 5;
  const int t0 = (w >> 3) * 16;        // token tile
  const int hh = w & 7;                // head tile (64 cols)
  const int c0 = hh * DD;
  v8f acc[4];
  #pragma unroll
  for (int ct = 0; ct < 4; ++ct) { v8f z = {}; acc[ct] = z; }
  for (int ks = 0; ks < HH / 32; ++ks) {
    Frag a = load_a_f32(X + (size_t)t0 * HH + ks * 32, HH, lane);
    #pragma unroll
    for (int ct = 0; ct < 4; ++ct) {
      Frag b = load_b_bf16(Wb + (size_t)(c0 + 16 * ct) * HH + ks * 32, HH, lane);
      acc[ct] = wmma_bf16(a, b, acc[ct]);
    }
  }
  const int i0 = 8 * (lane >> 4), j = lane & 15;
  #pragma unroll
  for (int ct = 0; ct < 4; ++ct) {
    #pragma unroll
    for (int r = 0; r < 8; ++r) {
      int tok = t0 + i0 + r;
      int b = tok / SEQ, s = tok % SEQ;
      int dd = 16 * ct + j;
      unsigned short val = f2bf(acc[ct][r]);
      if (MODE == 0)      dst[((size_t)(b * KK + hh) * MM + s) * DD + dd] = val;
      else if (MODE == 1) dst[((size_t)(b * KK + hh) * NP + s) * DD + dd] = val;
      else                dst[((size_t)(b * KK + hh) * DD + dd) * NP + s] = val;
    }
  }
}

// ============ attn: 4 waves/block, one wave per (head, 16-row m-tile) ============
__global__ void __launch_bounds__(128) attn_kernel(const unsigned short* __restrict__ ws,
                                                   unsigned short* __restrict__ att) {
  extern __shared__ char smem[];
  const int lane = threadIdx.x & 31;
  const int wv = threadIdx.x >> 5;                 // wave in block (0..3)
  char* mybase = smem + (size_t)wv * LDS_PER_WAVE;
  unsigned short* qpe = (unsigned short*)mybase;                 // 16x1024 bf16
  unsigned short* scp = (unsigned short*)(mybase + LDS_QPE_B);   // 16x1056 bf16: scores then p

  const int gw = blockIdx.x * 4 + wv;              // global wave-tile
  const int head = gw >> 5;                        // 64 heads
  const int m0 = (gw & 31) * 16;

  const unsigned short* qb = ws + WS_QB + (size_t)head * MM * DD;
  const unsigned short* kb = ws + WS_KB + (size_t)head * NP * DD;
  const unsigned short* vt = ws + WS_VT + (size_t)head * DD * NP;
  const unsigned short* pe = ws + WS_PE;

  // q fragments for this m-tile, kept live the whole kernel (K = D = 64 -> 2 frags)
  Frag a0 = load_a_bf16(qb + (size_t)m0 * DD,      DD, lane);
  Frag a1 = load_a_bf16(qb + (size_t)m0 * DD + 32, DD, lane);

  const int i0 = 8 * (lane >> 4), jj = lane & 15;

  // ---- Phase 1: qpe[m][l] = q . peT  (16 x 1024, WMMA, stored bf16 in LDS) ----
  for (int lt = 0; lt < LL / 16; ++lt) {
    v8f acc = {};
    Frag b0 = load_b_bf16(pe + (size_t)lt * 16 * DD,      DD, lane);
    Frag b1 = load_b_bf16(pe + (size_t)lt * 16 * DD + 32, DD, lane);
    acc = wmma_bf16(a0, b0, acc);
    acc = wmma_bf16(a1, b1, acc);
    #pragma unroll
    for (int r = 0; r < 8; ++r)
      qpe[(i0 + r) * LL + lt * 16 + jj] = f2bf(acc[r]);
  }
  __syncthreads();

  // ---- Phase 2: content scores over absolute n, + relative qpe gather, mask ----
  for (int jt = 0; jt < 66; ++jt) {
    int n0 = m0 + jt * 16;
    __builtin_prefetch(kb + (size_t)(n0 + 16) * DD + lane * 8, 0, 0);
    v8f acc = {};
    Frag b0 = load_b_bf16(kb + (size_t)n0 * DD,      DD, lane);
    Frag b1 = load_b_bf16(kb + (size_t)n0 * DD + 32, DD, lane);
    acc = wmma_bf16(a0, b0, acc);
    acc = wmma_bf16(a1, b1, acc);
    #pragma unroll
    for (int r = 0; r < 8; ++r) {
      int i = i0 + r;
      int l = jt * 16 + jj - i;              // relative offset = n - m
      float s = -1e30f;
      if ((unsigned)l < (unsigned)LL)
        s = (acc[r] + bf2f(qpe[i * LL + l])) * 0.125f;   // 1/sqrt(D)
      scp[i * 1056 + jt * 16 + jj] = f2bf(s);
    }
  }
  __syncthreads();

  // ---- Phase 3: softmax over 1056 cols (2 lanes per row), vectorized 8 bf16/load;
  //      p = exp(s - max) written back in place (unnormalized); 1/sum folded at the end
  const int row = lane & 15, half = lane >> 4;
  unsigned short* prow = scp + row * 1056 + half * 528;
  float mx = -1e30f;
  for (int t = 0; t < 66; ++t) {
    uint4 x = *(const uint4*)(prow + t * 8);
    unsigned int uu[4] = {x.x, x.y, x.z, x.w};
    #pragma unroll
    for (int k = 0; k < 4; ++k) {
      mx = fmaxf(mx, __uint_as_float(uu[k] << 16));
      mx = fmaxf(mx, __uint_as_float(uu[k] & 0xffff0000u));
    }
  }
  mx = fmaxf(mx, __shfl_xor(mx, 16));
  float sum = 0.f;
  for (int t = 0; t < 66; ++t) {
    uint4 x = *(const uint4*)(prow + t * 8);
    unsigned int uu[4] = {x.x, x.y, x.z, x.w};
    uint4 y;
    unsigned int yy[4];
    #pragma unroll
    for (int k = 0; k < 4; ++k) {
      float e0 = __expf(__uint_as_float(uu[k] << 16) - mx);
      float e1 = __expf(__uint_as_float(uu[k] & 0xffff0000u) - mx);
      sum += e0 + e1;
      yy[k] = ((unsigned int)f2bf(e1) << 16) | (unsigned int)f2bf(e0);
    }
    y.x = yy[0]; y.y = yy[1]; y.z = yy[2]; y.w = yy[3];
    *(uint4*)(prow + t * 8) = y;
  }
  sum += __shfl_xor(sum, 16);
  float rowinv = 1.0f / sum;
  __syncthreads();

  // ---- Phase 4: out[16x64] = p @ v  (33 K-steps of 32, B-frags from transposed v) ----
  v8f acc[4];
  #pragma unroll
  for (int ct = 0; ct < 4; ++ct) { v8f z = {}; acc[ct] = z; }
  for (int step = 0; step < 33; ++step) {
    __builtin_prefetch(vt + (size_t)(lane & 15) * NP + m0 + (step + 1) * 32, 0, 0);
    Frag a = load_a_bf16(scp + step * 32, 1056, lane);
    #pragma unroll
    for (int ct = 0; ct < 4; ++ct) {
      Frag b = load_b_bf16(vt + (size_t)(16 * ct) * NP + m0 + step * 32, NP, lane);
      acc[ct] = wmma_bf16(a, b, acc[ct]);
    }
  }
  const int b = head >> 3, kk = head & 7;
  #pragma unroll
  for (int ct = 0; ct < 4; ++ct) {
    #pragma unroll
    for (int r = 0; r < 8; ++r) {
      int i = i0 + r;
      float inv = __shfl(rowinv, i);
      att[((size_t)(b * MM + m0 + i)) * HH + kk * DD + 16 * ct + jj] =
          f2bf(acc[ct][r] * inv);
    }
  }
}

// ============ outproj: d_out = att @ Wo^T (fp32 result) ============
__global__ void __launch_bounds__(128) outproj_kernel(const unsigned short* __restrict__ ws,
                                                      float* __restrict__ out) {
  const int lane = threadIdx.x & 31;
  const int w = (blockIdx.x * blockDim.x + threadIdx.x) >> 5;
  const int t0 = (w >> 3) * 16;
  const int c0 = (w & 7) * DD;
  const unsigned short* att = ws + WS_ATT;
  const unsigned short* Wo  = ws + WS_WO;
  v8f acc[4];
  #pragma unroll
  for (int ct = 0; ct < 4; ++ct) { v8f z = {}; acc[ct] = z; }
  for (int ks = 0; ks < HH / 32; ++ks) {
    Frag a = load_a_bf16(att + (size_t)t0 * HH + ks * 32, HH, lane);
    #pragma unroll
    for (int ct = 0; ct < 4; ++ct) {
      Frag b = load_b_bf16(Wo + (size_t)(c0 + 16 * ct) * HH + ks * 32, HH, lane);
      acc[ct] = wmma_bf16(a, b, acc[ct]);
    }
  }
  const int i0 = 8 * (lane >> 4), j = lane & 15;
  #pragma unroll
  for (int ct = 0; ct < 4; ++ct)
    #pragma unroll
    for (int r = 0; r < 8; ++r)
      out[(size_t)(t0 + i0 + r) * HH + c0 + 16 * ct + j] = acc[ct][r];
}

extern "C" void kernel_launch(void* const* d_in, const int* in_sizes, int n_in,
                              void* d_out, int out_size, void* d_ws, size_t ws_size,
                              hipStream_t stream) {
  const float* query  = (const float*)d_in[0];
  const float* key    = (const float*)d_in[1];
  const float* value  = (const float*)d_in[2];
  const float* key_pe = (const float*)d_in[3];
  const float* Wq     = (const float*)d_in[4];
  const float* Wk     = (const float*)d_in[5];
  const float* Wv     = (const float*)d_in[6];
  const float* Wo     = (const float*)d_in[7];
  unsigned short* ws = (unsigned short*)d_ws;   // requires ws_size >= ~36.1 MB
  float* out = (float*)d_out;

  // prep: 4*262144 + 65536 + 2*65536 = 1,245,184 threads
  prep_kernel<<<4864, 256, 0, stream>>>(Wq, Wk, Wv, Wo, key_pe, ws);

  // projections: wave tile = 16 tokens x 64 cols; 4 waves / 128-thread block
  proj_kernel<0, MM><<<512,  128, 0, stream>>>(query, ws + WS_WQ, ws + WS_QB);
  proj_kernel<1, NN><<<1536, 128, 0, stream>>>(key,   ws + WS_WK, ws + WS_KB);
  proj_kernel<2, NN><<<1536, 128, 0, stream>>>(value, ws + WS_WV, ws + WS_VT);

  // attention: 64 heads * 32 m-tiles; 4 wave-tiles per 128-thread block
  size_t sm = (size_t)4 * LDS_PER_WAVE;   // 266,240 B dynamic LDS (< 320 KB/WGP)
  attn_kernel<<<512, 128, sm, stream>>>(ws, ws + WS_ATT);

  // output projection: 4096 rows x 512 cols
  outproj_kernel<<<512, 128, 0, stream>>>(ws, out);
}
